// TopKSAE_16569983828624
// MI455X (gfx1250) — matrix operand order repaired and checked by reading, not compiled, not run on hardware
//
#include <hip/hip_runtime.h>

// ---------------------------------------------------------------------------
// TopK SAE forward for MI455X (gfx1250, wave32, WMMA).
//   pre  = relu((x - b_dec) @ W_enc + b_enc)      [8192 x 16384]
//   z    = top-32-per-row sparsified pre
//   xhat = z @ W_dec + b_dec                      [8192 x 768]
// d_out = [ xhat | z ] (f32, concatenated).
// Encoder GEMM uses v_wmma_f32_16x16x32_bf16 with a hi/lo bf16 split
// (3 products) for near-f32 accuracy at bf16 matrix-pipe rate.
// ---------------------------------------------------------------------------

typedef __attribute__((ext_vector_type(16))) __bf16 v16bf;
typedef __attribute__((ext_vector_type(8)))  float  v8f;

#define BATCH 8192
#define DIN   768
#define DSAE  16384
#define KSEL  32

#define BM 128
#define BN 128
#define BK 32
#define LDH 40   // padded LDS row stride in halfwords (80B: 16B-aligned, conflict-friendly)

static __device__ __forceinline__ unsigned short f2bf(float f) {
  unsigned u = __float_as_uint(f);
  u += 0x7FFFu + ((u >> 16) & 1u);       // round-to-nearest-even
  return (unsigned short)(u >> 16);
}
static __device__ __forceinline__ float bf2f(unsigned short s) {
  return __uint_as_float(((unsigned)s) << 16);
}

union Frag { v16bf v; uint4 u[2]; };

// ---------------------------------------------------------------------------
// Kernel 1: encoder GEMM (bf16x3 split) + bias + relu -> z (dense pre-acts)
// grid (DSAE/BN, BATCH/BM), block 256 (8 waves); wave grid 2(M) x 4(N),
// each wave computes 64x32 = eight 16x16 f32 accumulators.
// ---------------------------------------------------------------------------
__global__ __launch_bounds__(256) void sae_enc_gemm(
    const float* __restrict__ x, const float* __restrict__ Wenc,
    const float* __restrict__ benc, const float* __restrict__ bdec,
    float* __restrict__ z)
{
  __shared__ __align__(16) unsigned short sAH[BM * LDH];
  __shared__ __align__(16) unsigned short sAL[BM * LDH];
  __shared__ __align__(16) unsigned short sBH[BN * LDH];
  __shared__ __align__(16) unsigned short sBL[BN * LDH];

  const int tid  = threadIdx.x;
  const int lane = tid & 31;
  const int wave = tid >> 5;
  const int wm   = wave >> 2;        // 0..1
  const int wn   = wave & 3;         // 0..3
  const int m0   = blockIdx.y * BM;
  const int n0   = blockIdx.x * BN;

  v8f acc[4][2];
  {
    v8f zf = {0.f, 0.f, 0.f, 0.f, 0.f, 0.f, 0.f, 0.f};
    for (int i = 0; i < 4; ++i)
      for (int j = 0; j < 2; ++j) acc[i][j] = zf;
  }

  for (int kt = 0; kt < DIN; kt += BK) {
    __syncthreads();
    // ---- stage A tile: (x - b_dec)[m0:m0+128, kt:kt+32] -> hi/lo bf16 ----
    {
      int r = tid >> 3;              // 0..31
      const int q = (tid & 7) * 4;   // f32 quad within the 32-wide K slab
      for (int s = 0; s < 4; ++s, r += 32) {
        const float4 xv = *(const float4*)(x + (size_t)(m0 + r) * DIN + kt + q);
        const float4 bd = *(const float4*)(bdec + kt + q);
        float f[4] = {xv.x - bd.x, xv.y - bd.y, xv.z - bd.z, xv.w - bd.w};
        for (int i = 0; i < 4; ++i) {
          unsigned short h = f2bf(f[i]);
          unsigned short l = f2bf(f[i] - bf2f(h));
          sAH[r * LDH + q + i] = h;
          sAL[r * LDH + q + i] = l;
        }
      }
    }
    // ---- stage B tile: W_enc[kt:kt+32, n0:n0+128], transposed (N-major) ----
    {
      int kr = tid >> 5;             // 0..7
      const int c = lane * 4;        // col quad
      for (int s = 0; s < 4; ++s, kr += 8) {
        const float4 wv = *(const float4*)(Wenc + (size_t)(kt + kr) * DSAE + n0 + c);
        float f[4] = {wv.x, wv.y, wv.z, wv.w};
        for (int i = 0; i < 4; ++i) {
          unsigned short h = f2bf(f[i]);
          unsigned short l = f2bf(f[i] - bf2f(h));
          sBH[(c + i) * LDH + kr] = h;
          sBL[(c + i) * LDH + kr] = l;
        }
      }
    }
    __syncthreads();

    // ---- fragment loads (ds_load_b128) ----
    Frag aH[4], aL[4], bH[2], bL[2];
    const int mlane = lane & 15;
    const int ko = (lane >> 4) << 3;   // A: K halves interleave by 8
    for (int tm = 0; tm < 4; ++tm) {
      const unsigned short* pH = &sAH[(wm * 64 + tm * 16 + mlane) * LDH + ko];
      const unsigned short* pL = &sAL[(wm * 64 + tm * 16 + mlane) * LDH + ko];
      aH[tm].u[0] = *(const uint4*)(pH);
      aH[tm].u[1] = *(const uint4*)(pH + 16);
      aL[tm].u[0] = *(const uint4*)(pL);
      aL[tm].u[1] = *(const uint4*)(pL + 16);
    }
    const int kb = (lane >> 4) << 3;   // B: K halves interleave by 8
    for (int tn = 0; tn < 2; ++tn) {
      const unsigned short* pH = &sBH[(wn * 32 + tn * 16 + mlane) * LDH + kb];
      const unsigned short* pL = &sBL[(wn * 32 + tn * 16 + mlane) * LDH + kb];
      bH[tn].u[0] = *(const uint4*)(pH);
      bH[tn].u[1] = *(const uint4*)(pH + 16);
      bL[tn].u[0] = *(const uint4*)(pL);
      bL[tn].u[1] = *(const uint4*)(pL + 16);
    }

    // ---- bf16x3 split WMMA: C += Ah*Bh + Ah*Bl + Al*Bh ----
    for (int tm = 0; tm < 4; ++tm)
      for (int tn = 0; tn < 2; ++tn) {
        acc[tm][tn] = __builtin_amdgcn_wmma_f32_16x16x32_bf16(
            false, aH[tm].v, false, bH[tn].v, (short)0, acc[tm][tn], false, false);
        acc[tm][tn] = __builtin_amdgcn_wmma_f32_16x16x32_bf16(
            false, aH[tm].v, false, bL[tn].v, (short)0, acc[tm][tn], false, false);
        acc[tm][tn] = __builtin_amdgcn_wmma_f32_16x16x32_bf16(
            false, aL[tm].v, false, bH[tn].v, (short)0, acc[tm][tn], false, false);
      }
  }

  // ---- epilogue: + b_enc, relu, store pre-activations into z ----
  const int mlane = lane & 15;
  const int moff = (lane >> 4) << 3;   // C layout: lanes 16..31 hold M+8
  for (int tm = 0; tm < 4; ++tm)
    for (int tn = 0; tn < 2; ++tn) {
      const int col = n0 + wn * 32 + tn * 16 + mlane;
      const float be = benc[col];
      const int rbase = m0 + wm * 64 + tm * 16 + moff;
      for (int r = 0; r < 8; ++r) {
        float v = acc[tm][tn][r] + be;
        v = v > 0.f ? v : 0.f;
        z[(size_t)(rbase + r) * DSAE + col] = v;
      }
    }
}

// ---------------------------------------------------------------------------
// Kernel 2: exact top-32 per row via MSB-first 8-bit radix select in LDS.
// relu'd values are >= 0, so uint bit order == float order. Rewrites the row
// (kept values, zeros elsewhere) and emits compact (val, idx) lists to d_ws.
// ---------------------------------------------------------------------------
__global__ __launch_bounds__(256) void sae_topk(
    float* __restrict__ z, float* __restrict__ cvals, int* __restrict__ cidx)
{
  extern __shared__ unsigned rowbits[];      // 16384 u32 (64 KB dynamic LDS)
  __shared__ unsigned hist[256];
  __shared__ int sGt[256], sEq[256];
  __shared__ unsigned sPrefix, sNeed;

  const int tid = threadIdx.x;
  const size_t row = blockIdx.x;
  float* zr = z + row * DSAE;

  {  // stage the row (vectorized, coalesced)
    const uint4* src = (const uint4*)zr;
    uint4* dst = (uint4*)rowbits;
    for (int i = tid; i < DSAE / 4; i += 256) dst[i] = src[i];
  }
  if (tid == 0) { sPrefix = 0u; sNeed = KSEL; }

  const unsigned premask[4] = {0u, 0xFF000000u, 0xFFFF0000u, 0xFFFFFF00u};
  for (int pass = 0; pass < 4; ++pass) {
    hist[tid] = 0u;
    __syncthreads();
    const int shift = 24 - 8 * pass;
    const unsigned pm = premask[pass];
    const unsigned pref = sPrefix;
    for (int i = tid; i < DSAE; i += 256) {
      const unsigned u = rowbits[i];
      if ((u & pm) == (pref & pm)) atomicAdd(&hist[(u >> shift) & 0xFFu], 1u);
    }
    __syncthreads();
    if (tid == 0) {
      unsigned need = sNeed, cum = 0;
      for (int b = 255; b >= 0; --b) {
        const unsigned h = hist[b];
        if (cum + h >= need) {
          sPrefix = pref | ((unsigned)b << shift);
          sNeed = need - cum;
          break;
        }
        cum += h;
      }
    }
    __syncthreads();
  }

  const unsigned T = sPrefix;       // exact 32nd-largest bit pattern
  const int tieB = (int)sNeed;      // ties at T still to keep
  const int g = KSEL - tieB;        // count strictly greater than T

  int cgt = 0, ceq = 0;
  for (int i = tid; i < DSAE; i += 256) {
    const unsigned u = rowbits[i];
    cgt += (u > T);
    ceq += (u == T);
  }
  sGt[tid] = cgt; sEq[tid] = ceq;
  __syncthreads();
  if (tid == 0) {                   // deterministic exclusive scans
    int r1 = 0, r2 = 0;
    for (int t = 0; t < 256; ++t) {
      const int a = sGt[t]; sGt[t] = r1; r1 += a;
      const int b = sEq[t]; sEq[t] = r2; r2 += b;
    }
  }
  __syncthreads();

  int gslot = sGt[tid], eslot = sEq[tid];
  for (int i = tid; i < DSAE; i += 256) {
    const unsigned u = rowbits[i];
    const float v = __uint_as_float(u);
    float outv = 0.0f;
    if (u > T) {
      const int s = gslot++;
      cvals[row * KSEL + s] = v; cidx[row * KSEL + s] = i;
      outv = v;
    } else if (u == T) {
      const int e = eslot++;
      if (e < tieB) {
        const int s = g + e;
        cvals[row * KSEL + s] = v; cidx[row * KSEL + s] = i;
        outv = v;
      }
    }
    zr[i] = outv;                   // coalesced full-row rewrite
  }
}

// ---------------------------------------------------------------------------
// Kernel 3: sparse decoder. x_hat[row] = b_dec + sum_i cval_i * W_dec[cidx_i].
// W_dec (48 MB) is resident in the 192 MB L2, so the gathers are cheap.
// ---------------------------------------------------------------------------
__global__ __launch_bounds__(192) void sae_decode(
    const float* __restrict__ Wdec, const float* __restrict__ bdec,
    const float* __restrict__ cvals, const int* __restrict__ cidx,
    float* __restrict__ xhat)
{
  __shared__ float sv[KSEL];
  __shared__ int   si[KSEL];
  const size_t row = blockIdx.x;
  if (threadIdx.x < KSEL) {
    sv[threadIdx.x] = cvals[row * KSEL + threadIdx.x];
    si[threadIdx.x] = cidx[row * KSEL + threadIdx.x];
  }
  __syncthreads();

  const int j = threadIdx.x;        // 192 threads x float4 = 768 cols
  float4 acc = ((const float4*)bdec)[j];
  for (int i = 0; i < KSEL; ++i) {
    const float v = sv[i];
    if (v != 0.f) {
      const float4 w = ((const float4*)(Wdec + (size_t)si[i] * DIN))[j];
      acc.x += v * w.x; acc.y += v * w.y; acc.z += v * w.z; acc.w += v * w.w;
    }
  }
  ((float4*)(xhat + row * DIN))[j] = acc;
}

// ---------------------------------------------------------------------------
extern "C" void kernel_launch(void* const* d_in, const int* in_sizes, int n_in,
                              void* d_out, int out_size, void* d_ws, size_t ws_size,
                              hipStream_t stream) {
  (void)in_sizes; (void)n_in; (void)out_size; (void)ws_size;
  const float* x     = (const float*)d_in[0];
  const float* W_enc = (const float*)d_in[1];
  const float* b_enc = (const float*)d_in[2];
  const float* W_dec = (const float*)d_in[3];
  const float* b_dec = (const float*)d_in[4];
  // d_in[5] is k (== 32, compile-time constant in the reference module)

  float* xhat = (float*)d_out;
  float* z    = (float*)d_out + (size_t)BATCH * DIN;
  float* cvals = (float*)d_ws;                                   // 1 MB
  int*   cidx  = (int*)((char*)d_ws + (size_t)BATCH * KSEL * 4); // 1 MB

  dim3 ggrid(DSAE / BN, BATCH / BM);
  sae_enc_gemm<<<ggrid, 256, 0, stream>>>(x, W_enc, b_enc, b_dec, z);
  sae_topk<<<BATCH, 256, DSAE * sizeof(unsigned), stream>>>(z, cvals, cidx);
  sae_decode<<<BATCH, 192, 0, stream>>>(W_dec, b_dec, cvals, cidx, xhat);
}
